// ATL_31430570672603
// MI455X (gfx1250) — matrix operand order repaired and checked by reading, not compiled, not run on hardware
//
#include <hip/hip_runtime.h>
#include <hip/hip_bf16.h>
#include <math.h>

typedef __attribute__((ext_vector_type(16))) _Float16 v16h;
typedef __attribute__((ext_vector_type(8)))  _Float16 v8h;
typedef __attribute__((ext_vector_type(4)))  _Float16 v4h;
typedef __attribute__((ext_vector_type(8)))  float    v8f;

union HalfFrag { v16h v; v8h h[2]; };

#define WAVE_M 4   // 16x16 tiles per wave in M
#define WAVE_N 4   // 16x16 tiles per wave in N
// wave tile 64x64, workgroup = 4 waves (2x2) -> 128x128 macro tile

// ---------------------------------------------------------------------------
// f32 -> f16 convert, 4 elements / thread (b128 load, b64 store)
// ---------------------------------------------------------------------------
__global__ __launch_bounds__(256)
void ATL_cvt_f16(const float* __restrict__ in, _Float16* __restrict__ out, int n4) {
  int i = blockIdx.x * 256 + threadIdx.x;
  if (i >= n4) return;
  float4 v = ((const float4*)in)[i];
  v4h h;
  h[0] = (_Float16)v.x; h[1] = (_Float16)v.y;
  h[2] = (_Float16)v.z; h[3] = (_Float16)v.w;
  ((v4h*)out)[i] = h;
}

// ---------------------------------------------------------------------------
// C[M,N](f32) = A[M,K](f16, row-major) * B[N,K](f16, row-major)^T  (+ bias[N])
// WMMA f32_16x16x32_f16. Fragment layouts per CDNA5 ISA 7.12.2:
//   A 16x32: lanes 0-15 row=lane, halves K {0..7,16..23}; lanes 16-31 K {8..15,24..31}
//   B 32x16: lanes 0-15 col=lane, halves K {0..7,16..23}; lanes 16-31 K {8..15,24..31}
//   C 16x16: lane n=lane&15, VGPR r -> row (lane>=16 ? 8+r : r)
// All tile dims divide M,N,K exactly -> no divergence (EXEC all ones for WMMA).
// ---------------------------------------------------------------------------
__global__ __launch_bounds__(128)
void ATL_gemm_f16nt(const _Float16* __restrict__ A,
                    const _Float16* __restrict__ Bm,
                    float* __restrict__ C,
                    const float* __restrict__ bias,
                    int M, int N, int K)
{
  const int lane  = threadIdx.x & 31;
  const int wave  = threadIdx.x >> 5;
  const int tileM0 = blockIdx.y * 128 + (wave >> 1) * (WAVE_M * 16);
  const int tileN0 = blockIdx.x * 128 + (wave & 1) * (WAVE_N * 16);

  const int rowA  = lane & 15;
  const int kOffA = (lane >> 4) << 3;   // 0 or 8
  const int rowB  = lane & 15;
  const int kOffB = (lane >> 4) << 3;   // 0 or 8

  v8f zero = {};
  v8f acc[WAVE_M][WAVE_N];
#pragma unroll
  for (int m = 0; m < WAVE_M; ++m)
#pragma unroll
    for (int n = 0; n < WAVE_N; ++n)
      acc[m][n] = zero;

  const _Float16* aBase = A  + (size_t)(tileM0 + rowA) * K + kOffA;
  const _Float16* bBase = Bm + (size_t)(tileN0 + rowB) * K + kOffB;

  for (int k0 = 0; k0 < K; k0 += 32) {
    HalfFrag aF[WAVE_M], bF[WAVE_N];
#pragma unroll
    for (int m = 0; m < WAVE_M; ++m) {
      const _Float16* p = aBase + (size_t)(m * 16) * K + k0;
      aF[m].h[0] = *(const v8h*)(p);        // K = kOffA .. kOffA+7
      aF[m].h[1] = *(const v8h*)(p + 16);   // K = kOffA+16 .. kOffA+23
    }
#pragma unroll
    for (int n = 0; n < WAVE_N; ++n) {
      const _Float16* p = bBase + (size_t)(n * 16) * K + k0;
      bF[n].h[0] = *(const v8h*)(p);        // K = kOffB .. kOffB+7
      bF[n].h[1] = *(const v8h*)(p + 16);   // K = kOffB+16 .. kOffB+23
    }
#pragma unroll
    for (int n = 0; n < WAVE_N; ++n)
#pragma unroll
      for (int m = 0; m < WAVE_M; ++m)
        acc[m][n] = __builtin_amdgcn_wmma_f32_16x16x32_f16(
            false, aF[m].v, false, bF[n].v, (short)0, acc[m][n], false, false);
  }

  const int colC  = lane & 15;
  const int mBase = (lane >> 4) << 3;
#pragma unroll
  for (int m = 0; m < WAVE_M; ++m) {
#pragma unroll
    for (int n = 0; n < WAVE_N; ++n) {
      const int col = tileN0 + n * 16 + colC;
      const float badd = bias ? bias[col] : 0.0f;
#pragma unroll
      for (int r = 0; r < 8; ++r) {
        const int row = tileM0 + m * 16 + mBase + r;
        C[(size_t)row * N + col] = acc[m][n][r] + badd;
      }
    }
  }
}

// ---------------------------------------------------------------------------
// L2-normalize rows of P[rows,D] (f32), emit f16 for next GEMM's A matrix.
// ---------------------------------------------------------------------------
__global__ __launch_bounds__(256)
void ATL_normalize(const float* __restrict__ P, _Float16* __restrict__ out, int D) {
  const int row = blockIdx.x;
  const float* p = P + (size_t)row * D;
  float ss = 0.0f;
  for (int i = threadIdx.x; i < D; i += 256) { float v = p[i]; ss += v * v; }
  __shared__ float red[256];
  red[threadIdx.x] = ss;
  __syncthreads();
  for (int off = 128; off > 0; off >>= 1) {
    if (threadIdx.x < off) red[threadIdx.x] += red[threadIdx.x + off];
    __syncthreads();
  }
  const float inv = 1.0f / fmaxf(sqrtf(red[0]), 1e-12f);
  _Float16* o = out + (size_t)row * D;
  for (int i = threadIdx.x; i < D; i += 256) o[i] = (_Float16)(p[i] * inv);
}

// ---------------------------------------------------------------------------
// One workgroup per row: online softmax stats (max logit, shifted sum) + argmax.
// logits = sim * 10 (TEMPERATURE = 0.1)
// ---------------------------------------------------------------------------
__global__ __launch_bounds__(256)
void ATL_row_stats(const float* __restrict__ S, float* __restrict__ rowMax,
                   float* __restrict__ rowSum, int* __restrict__ rowArg, int N) {
  const int row = blockIdx.x;
  const float* s = S + (size_t)row * N;
  float m = -INFINITY, sum = 0.0f, bestV = -INFINITY;
  int bestI = 0x7fffffff;
  for (int i = threadIdx.x; i < N; i += 256) {
    const float x = s[i];
    if (x > bestV) { bestV = x; bestI = i; }
    const float l = x * 10.0f;
    if (l > m) { sum = sum * __expf(m - l) + 1.0f; m = l; }
    else       { sum += __expf(l - m); }
  }
  __shared__ float sm[256], ssum[256], sbv[256];
  __shared__ int   sbi[256];
  const int t = threadIdx.x;
  sm[t] = m; ssum[t] = sum; sbv[t] = bestV; sbi[t] = bestI;
  __syncthreads();
  for (int off = 128; off > 0; off >>= 1) {
    if (t < off) {
      const float m1 = sm[t], m2 = sm[t + off];
      const float mm = fmaxf(m1, m2);
      ssum[t] = ssum[t] * __expf(m1 - mm) + ssum[t + off] * __expf(m2 - mm);
      sm[t] = mm;
      const float v2 = sbv[t + off]; const int i2 = sbi[t + off];
      if (v2 > sbv[t] || (v2 == sbv[t] && i2 < sbi[t])) { sbv[t] = v2; sbi[t] = i2; }
    }
    __syncthreads();
  }
  if (t == 0) { rowMax[row] = sm[0]; rowSum[row] = ssum[0]; rowArg[row] = sbi[0]; }
}

// ---------------------------------------------------------------------------
// activations = exp(sim*10 - rowMax) / rowSum, float4 vectorized.
// ---------------------------------------------------------------------------
__global__ __launch_bounds__(256)
void ATL_finalize(const float* __restrict__ S, const float* __restrict__ rowMax,
                  const float* __restrict__ rowSum, float* __restrict__ out, int N) {
  const int row = blockIdx.y;
  const int i = (blockIdx.x * 256 + threadIdx.x) * 4;
  const float m = rowMax[row];
  const float inv = 1.0f / rowSum[row];
  const float4 x = *(const float4*)(S + (size_t)row * N + i);
  float4 y;
  y.x = __expf(x.x * 10.0f - m) * inv;
  y.y = __expf(x.y * 10.0f - m) * inv;
  y.z = __expf(x.z * 10.0f - m) * inv;
  y.w = __expf(x.w * 10.0f - m) * inv;
  *(float4*)(out + (size_t)row * N + i) = y;
}

__global__ __launch_bounds__(256)
void ATL_write_idx(const int* __restrict__ idx, float* __restrict__ out, int n) {
  int i = blockIdx.x * 256 + threadIdx.x;
  if (i < n) out[i] = (float)idx[i];
}

// ---------------------------------------------------------------------------
extern "C" void kernel_launch(void* const* d_in, const int* in_sizes, int n_in,
                              void* d_out, int out_size, void* d_ws, size_t ws_size,
                              hipStream_t stream) {
  const float* features = (const float*)d_in[0];  // [B, D]
  const float* W        = (const float*)d_in[1];  // [D, D]
  const float* bias     = (const float*)d_in[2];  // [D]
  const float* protos   = (const float*)d_in[3];  // [Kc, D]

  const int D  = in_sizes[2];          // 1024
  const int B  = in_sizes[0] / D;      // 1024
  const int Kc = in_sizes[3] / D;      // 32768

  char* ws = (char*)d_ws;
  _Float16* protoH = (_Float16*)ws; ws += (size_t)Kc * D * sizeof(_Float16);
  _Float16* featH  = (_Float16*)ws; ws += (size_t)B  * D * sizeof(_Float16);
  _Float16* WH     = (_Float16*)ws; ws += (size_t)D  * D * sizeof(_Float16);
  float*    P      = (float*)ws;    ws += (size_t)B  * D * sizeof(float);
  _Float16* projH  = (_Float16*)ws; ws += (size_t)B  * D * sizeof(_Float16);
  float*    S      = (float*)ws;    ws += (size_t)B  * Kc * sizeof(float);
  float*    rowMax = (float*)ws;    ws += (size_t)B * sizeof(float);
  float*    rowSum = (float*)ws;    ws += (size_t)B * sizeof(float);
  int*      rowArg = (int*)ws;      ws += (size_t)B * sizeof(int);

  // 1) f32 -> f16 conversions
  {
    int n4 = (Kc * D) / 4;
    ATL_cvt_f16<<<(n4 + 255) / 256, 256, 0, stream>>>(protos, protoH, n4);
    n4 = (B * D) / 4;
    ATL_cvt_f16<<<(n4 + 255) / 256, 256, 0, stream>>>(features, featH, n4);
    n4 = (D * D) / 4;
    ATL_cvt_f16<<<(n4 + 255) / 256, 256, 0, stream>>>(W, WH, n4);
  }

  // 2) projection: P = featH * WH^T + bias    [B x D]
  ATL_gemm_f16nt<<<dim3(D / 128, B / 128), 128, 0, stream>>>(
      featH, WH, P, bias, B, D, D);

  // 3) row L2 normalize -> f16
  ATL_normalize<<<B, 256, 0, stream>>>(P, projH, D);

  // 4) similarities: S = projH * protoH^T     [B x Kc]
  ATL_gemm_f16nt<<<dim3(Kc / 128, B / 128), 128, 0, stream>>>(
      projH, protoH, S, nullptr, B, Kc, D);

  // 5) per-row online softmax stats + argmax
  ATL_row_stats<<<B, 256, 0, stream>>>(S, rowMax, rowSum, rowArg, Kc);

  // 6) activations + indices
  ATL_finalize<<<dim3(Kc / 1024, B), 256, 0, stream>>>(
      S, rowMax, rowSum, (float*)d_out, Kc);
  ATL_write_idx<<<(B + 255) / 256, 256, 0, stream>>>(
      rowArg, (float*)d_out + (size_t)B * Kc, B);
}
